// CrossTaskAttention_8873402433774
// MI455X (gfx1250) — matrix operand order repaired and checked by reading, not compiled, not run on hardware
//
#include <hip/hip_runtime.h>
#include <hip/hip_bf16.h>
#include <math.h>

#define HIDDEN 1024
#define NUM_HEADS 16
#define HEAD_DIM 64
#define NUM_LANGS 5
#define BATCH 4
#define SEQ 1024

typedef __attribute__((ext_vector_type(16))) __bf16 bf16x16;
typedef __attribute__((ext_vector_type(8)))  float  f32x8;

union BfFrag {
    bf16x16 v;
    uint4 q[2];
    unsigned short u[16];
};

__device__ __forceinline__ unsigned short f32_to_bf16(float f) {
    unsigned int x = __float_as_uint(f);
    x += 0x7FFFu + ((x >> 16) & 1u);   // round-to-nearest-even
    return (unsigned short)(x >> 16);
}

// A-matrix fragment (16 x 32 bf16), rows row0..row0+15, k window [kcol0, kcol0+32)
// lane L holds row M = row0 + (L%16); lanes 0-15: k {0..7,16..23}, lanes 16-31: k {8..15,24..31}
__device__ __forceinline__ bf16x16 ldfragA(const unsigned short* __restrict__ base,
                                           int stride, int row0, int kcol0, int lane) {
    int r  = row0 + (lane & 15);
    int kb = kcol0 + ((lane & 16) ? 8 : 0);
    const unsigned short* p = base + (size_t)r * stride + kb;
    BfFrag f;
    f.q[0] = *reinterpret_cast<const uint4*>(p);
    f.q[1] = *reinterpret_cast<const uint4*>(p + 16);
    return f.v;
}

// B-matrix fragment (32 x 16 bf16), cols col0..col0+15, k window [krow0, krow0+32)
// source is stored [n][k] (row-major over k) -> lane L holds col N = col0 + (L%16),
// lanes 0-15: k {0..7,16..23}, lanes 16-31: k {8..15,24..31}
__device__ __forceinline__ bf16x16 ldfragB(const unsigned short* __restrict__ base,
                                           int stride, int col0, int krow0, int lane) {
    int c  = col0 + (lane & 15);
    int kb = krow0 + ((lane & 16) ? 8 : 0);
    const unsigned short* p = base + (size_t)c * stride + kb;
    BfFrag f;
    f.q[0] = *reinterpret_cast<const uint4*>(p);
    f.q[1] = *reinterpret_cast<const uint4*>(p + 16);
    return f.v;
}

__device__ __forceinline__ f32x8 bwmma(bf16x16 a, bf16x16 b, f32x8 c) {
    return __builtin_amdgcn_wmma_f32_16x16x32_bf16(false, a, false, b, (short)0, c,
                                                   false, false);
}

// ---------------------------------------------------------------------------
// Projection: out[b,h,s,d] (bf16) = X[m,:] @ W[n,:]^T + bias[n] (+ lang bias)
// M = B*SEQ = 4096, N = HIDDEN = 1024. 64x64 block, 128 threads (4 waves, 2x2).
// ---------------------------------------------------------------------------
template <int KDIM, bool ADD_LANG>
__launch_bounds__(128)
__global__ void proj_kernel(const float* __restrict__ X,
                            const float* __restrict__ W,
                            const float* __restrict__ bias,
                            const float* __restrict__ langb,
                            const long long* __restrict__ lang_ids,
                            unsigned short* __restrict__ out) {
    __shared__ __align__(16) unsigned short As[64 * 40];  // padded stride 40
    __shared__ __align__(16) unsigned short Bs[64 * 40];

    const int t = threadIdx.x, lane = t & 31, w = t >> 5;
    const int n0 = blockIdx.x * 64;
    const int m0 = blockIdx.y * 64;
    const int bidx = m0 >> 10;            // 1024 rows per batch, tiles never straddle
    const int wm = w >> 1, wn = w & 1;

    f32x8 acc[2][2] = {};

    for (int k0 = 0; k0 < KDIM; k0 += 32) {
        // stage A (64x32) and B (64x32) tiles, f32 -> bf16
        #pragma unroll
        for (int i = 0; i < 4; ++i) {
            int c = t + i * 128;               // 0..511 float4 chunks
            int row = c >> 3;
            int col = (c & 7) * 4;
            float4 va = *reinterpret_cast<const float4*>(
                X + (size_t)(m0 + row) * KDIM + k0 + col);
            uint2 pa;
            pa.x = (unsigned)f32_to_bf16(va.x) | ((unsigned)f32_to_bf16(va.y) << 16);
            pa.y = (unsigned)f32_to_bf16(va.z) | ((unsigned)f32_to_bf16(va.w) << 16);
            *reinterpret_cast<uint2*>(&As[row * 40 + col]) = pa;

            float4 vb = *reinterpret_cast<const float4*>(
                W + (size_t)(n0 + row) * KDIM + k0 + col);
            uint2 pb;
            pb.x = (unsigned)f32_to_bf16(vb.x) | ((unsigned)f32_to_bf16(vb.y) << 16);
            pb.y = (unsigned)f32_to_bf16(vb.z) | ((unsigned)f32_to_bf16(vb.w) << 16);
            *reinterpret_cast<uint2*>(&Bs[row * 40 + col]) = pb;
        }
        __syncthreads();

        bf16x16 a0 = ldfragA(As, 40, wm * 32, 0, lane);
        bf16x16 a1 = ldfragA(As, 40, wm * 32 + 16, 0, lane);
        bf16x16 b0 = ldfragB(Bs, 40, wn * 32, 0, lane);
        bf16x16 b1 = ldfragB(Bs, 40, wn * 32 + 16, 0, lane);
        acc[0][0] = bwmma(a0, b0, acc[0][0]);
        acc[0][1] = bwmma(a0, b1, acc[0][1]);
        acc[1][0] = bwmma(a1, b0, acc[1][0]);
        acc[1][1] = bwmma(a1, b1, acc[1][1]);
        __syncthreads();
    }

    long long lid = ADD_LANG ? lang_ids[bidx] : 0;

    #pragma unroll
    for (int ni = 0; ni < 2; ++ni) {
        int n = n0 + wn * 32 + ni * 16 + (lane & 15);
        float bv = bias[n];
        if (ADD_LANG) bv += langb[lid * HIDDEN + n];
        int h = n >> 6, d = n & 63;
        #pragma unroll
        for (int mi = 0; mi < 2; ++mi) {
            #pragma unroll
            for (int i = 0; i < 8; ++i) {
                int m = m0 + wm * 32 + mi * 16 + ((lane & 16) ? 8 + i : i);
                int s = m & 1023;
                float val = acc[mi][ni][i] + bv;
                out[(((size_t)bidx * NUM_HEADS + h) * SEQ + s) * HEAD_DIM + d] =
                    f32_to_bf16(val);
            }
        }
    }
}

// ---------------------------------------------------------------------------
// Fused attention: per (b, h, 32-row q tile). 256 threads = 8 waves.
// LDS: Qs bf16[32][72] | Sc f32[32][1024] (reused as V^T bf16[64][1032]) |
//      Pn bf16[32][1032]   => 202,752 bytes dynamic LDS.
// ---------------------------------------------------------------------------
#define QS_BYTES   (32 * 72 * 2)                     // 4608
#define SCVT_BYTES 132096                            // max(32*1024*4, 64*1032*2)
#define PN_BYTES   (32 * 1032 * 2)                   // 66048
#define ATTN_SMEM  (QS_BYTES + SCVT_BYTES + PN_BYTES)

__launch_bounds__(256)
__global__ void attn_kernel(const unsigned short* __restrict__ Qw,
                            const unsigned short* __restrict__ Kw,
                            const unsigned short* __restrict__ Vw,
                            const int* __restrict__ mask,
                            float* __restrict__ ctx,
                            float* __restrict__ attn) {
    extern __shared__ char smem[];
    unsigned short* Qs  = reinterpret_cast<unsigned short*>(smem);              // stride 72
    float*          Sc  = reinterpret_cast<float*>(smem + QS_BYTES);            // stride 1024
    unsigned short* VTs = reinterpret_cast<unsigned short*>(smem + QS_BYTES);   // stride 1032
    unsigned short* Pn  = reinterpret_cast<unsigned short*>(smem + QS_BYTES + SCVT_BYTES);

    const int t = threadIdx.x, lane = t & 31, w = t >> 5;
    const int q0 = blockIdx.x * 32, h = blockIdx.y, b = blockIdx.z;
    const size_t bh = (size_t)(b * NUM_HEADS + h);
    const unsigned short* Qbh = Qw + bh * SEQ * HEAD_DIM;
    const unsigned short* Kbh = Kw + bh * SEQ * HEAD_DIM;
    const unsigned short* Vbh = Vw + bh * SEQ * HEAD_DIM;

    // ---- load Q tile (32 x 64 bf16), 8 elements per thread ----
    {
        int idx = t * 8;
        int r = idx >> 6, c = idx & 63;
        *reinterpret_cast<uint4*>(&Qs[r * 72 + c]) =
            *reinterpret_cast<const uint4*>(Qbh + (size_t)(q0 + r) * HEAD_DIM + c);
    }
    __syncthreads();

    // ---- phase 1: S = (Q K^T) * 0.125, masked, into LDS ----
    bf16x16 afr[2][2];
    afr[0][0] = ldfragA(Qs, 72, 0, 0, lane);
    afr[0][1] = ldfragA(Qs, 72, 0, 32, lane);
    afr[1][0] = ldfragA(Qs, 72, 16, 0, lane);
    afr[1][1] = ldfragA(Qs, 72, 16, 32, lane);

    for (int nblk = 0; nblk < 8; ++nblk) {
        int n0 = w * 128 + nblk * 16;
        int n = n0 + (lane & 15);
        // K fragment straight from global: lane needs K[n][d0..d0+7] and K[n][d0+16..d0+23] (two 16B loads)
        bf16x16 bf0 = ldfragB(Kbh, HEAD_DIM, n0, 0, lane);
        bf16x16 bf1 = ldfragB(Kbh, HEAD_DIM, n0, 32, lane);
        int mv = mask[b * SEQ + n];
        #pragma unroll
        for (int mi = 0; mi < 2; ++mi) {
            f32x8 acc = {};
            acc = bwmma(afr[mi][0], bf0, acc);
            acc = bwmma(afr[mi][1], bf1, acc);
            #pragma unroll
            for (int i = 0; i < 8; ++i) {
                int row = mi * 16 + ((lane & 16) ? 8 + i : i);
                float val = acc[i] * 0.125f;
                if (mv == 0) val = -INFINITY;
                Sc[row * 1024 + n] = val;
            }
        }
    }
    __syncthreads();

    // ---- phase 2: row softmax; write attn (global, once) + bf16 P to LDS ----
    {
        int row = t >> 3, part = t & 7;           // 8 threads per row, 128 cols each
        float* rp = Sc + row * 1024 + part * 128;
        float mx = -INFINITY;
        for (int j = 0; j < 128; j += 4) {
            float4 v = *reinterpret_cast<float4*>(rp + j);
            mx = fmaxf(mx, fmaxf(fmaxf(v.x, v.y), fmaxf(v.z, v.w)));
        }
        #pragma unroll
        for (int off = 1; off < 8; off <<= 1) mx = fmaxf(mx, __shfl_xor(mx, off));
        float sum = 0.f;
        for (int j = 0; j < 128; j += 4) {
            float4 v = *reinterpret_cast<float4*>(rp + j);
            v.x = __expf(v.x - mx); v.y = __expf(v.y - mx);
            v.z = __expf(v.z - mx); v.w = __expf(v.w - mx);
            sum += v.x + v.y + v.z + v.w;
            *reinterpret_cast<float4*>(rp + j) = v;
        }
        #pragma unroll
        for (int off = 1; off < 8; off <<= 1) sum += __shfl_xor(sum, off);
        float inv = 1.0f / sum;
        float* gp = attn + (bh * SEQ + (q0 + row)) * SEQ + part * 128;
        unsigned short* pp = Pn + row * 1032 + part * 128;
        for (int j = 0; j < 128; j += 4) {
            float4 v = *reinterpret_cast<float4*>(rp + j);
            v.x *= inv; v.y *= inv; v.z *= inv; v.w *= inv;
            *reinterpret_cast<float4*>(gp + j) = v;   // coalesced f32 attn output
            uint2 pk;
            pk.x = (unsigned)f32_to_bf16(v.x) | ((unsigned)f32_to_bf16(v.y) << 16);
            pk.y = (unsigned)f32_to_bf16(v.z) | ((unsigned)f32_to_bf16(v.w) << 16);
            *reinterpret_cast<uint2*>(pp + j) = pk;
        }
    }
    __syncthreads();   // Sc is dead now; reuse as V^T

    // ---- phase 2.5: stage V^T (64 d x 1024 kv, bf16) into LDS ----
    for (int kv = t; kv < SEQ; kv += 256) {
        const unsigned short* vr = Vbh + (size_t)kv * HEAD_DIM;
        #pragma unroll
        for (int dq = 0; dq < 8; ++dq) {
            union { uint4 q; unsigned short u[8]; } blk;
            blk.q = *reinterpret_cast<const uint4*>(vr + dq * 8);
            #pragma unroll
            for (int j = 0; j < 8; ++j)
                VTs[(dq * 8 + j) * 1032 + kv] = blk.u[j];
        }
    }
    __syncthreads();

    // ---- phase 3: ctx = P @ V; each wave owns one 16x16 output frag ----
    {
        int mi = w >> 2, ni = w & 3;
        f32x8 acc = {};
        for (int k0 = 0; k0 < SEQ; k0 += 32) {
            bf16x16 a  = ldfragA(Pn, 1032, mi * 16, k0, lane);
            bf16x16 bb = ldfragB(VTs, 1032, ni * 16, k0, lane);
            acc = bwmma(a, bb, acc);
        }
        int d = ni * 16 + (lane & 15);
        size_t col = (size_t)h * HEAD_DIM + d;
        #pragma unroll
        for (int i = 0; i < 8; ++i) {
            int q = q0 + mi * 16 + ((lane & 16) ? 8 + i : i);
            ctx[((size_t)b * SEQ + q) * HIDDEN + col] = acc[i];
        }
    }
}

// ---------------------------------------------------------------------------
extern "C" void kernel_launch(void* const* d_in, const int* in_sizes, int n_in,
                              void* d_out, int out_size, void* d_ws, size_t ws_size,
                              hipStream_t stream) {
    const float*     query    = (const float*)d_in[0];
    const float*     key      = (const float*)d_in[1];
    const float*     value    = (const float*)d_in[2];
    const long long* lang_ids = (const long long*)d_in[3];   // int64
    const int*       mask     = (const int*)d_in[4];         // int32
    const float*     Wq = (const float*)d_in[5];
    const float*     bq = (const float*)d_in[6];
    const float*     Wk = (const float*)d_in[7];
    const float*     bk = (const float*)d_in[8];
    const float*     Wv = (const float*)d_in[9];
    const float*     bv = (const float*)d_in[10];
    const float*     lb = (const float*)d_in[11];

    // workspace: Q/K/V in bf16, [B,H,S,D] each = 8 MB (24 MB total)
    const size_t slab = (size_t)BATCH * NUM_HEADS * SEQ * HEAD_DIM;
    unsigned short* Qws = (unsigned short*)d_ws;
    unsigned short* Kws = Qws + slab;
    unsigned short* Vws = Kws + slab;

    dim3 pgrid(HIDDEN / 64, (BATCH * SEQ) / 64);   // (16, 64)
    proj_kernel<768, true ><<<pgrid, 128, 0, stream>>>(query, Wq, bq, lb, lang_ids, Qws);
    proj_kernel<512, false><<<pgrid, 128, 0, stream>>>(key,   Wk, bk, nullptr, nullptr, Kws);
    proj_kernel<512, false><<<pgrid, 128, 0, stream>>>(value, Wv, bv, nullptr, nullptr, Vws);

    float* ctx  = (float*)d_out;                                  // [B,SQ,HIDDEN]
    float* attn = ctx + (size_t)BATCH * SEQ * HIDDEN;             // [B,H,SQ,SKV]

    hipFuncSetAttribute((const void*)attn_kernel,
                        hipFuncAttributeMaxDynamicSharedMemorySize, ATTN_SMEM);
    attn_kernel<<<dim3(SEQ / 32, NUM_HEADS, BATCH), 256, ATTN_SMEM, stream>>>(
        Qws, Kws, Vws, mask, ctx, attn);
}
